// MyModelwoM_48533130445498
// MI455X (gfx1250) — compile-verified, matrix operand order checked by reading, not run on hardware
//
#include <hip/hip_runtime.h>
#include <hip/hip_bf16.h>
#include <math.h>

#define S_SNAP 8
#define N_TOT  20000
#define NSUB   4000
#define F_IN   128
#define HID    64
#define NH     8
#define DH     (HID / NH)

typedef __attribute__((ext_vector_type(16))) _Float16 v16h;
typedef __attribute__((ext_vector_type(8)))  _Float16 v8h;
typedef __attribute__((ext_vector_type(8)))  float    v8f;
typedef __attribute__((ext_vector_type(4)))  float    f32x4;

// Load 8 consecutive f32 (16B-aligned), optional non-temporal, convert to 8 f16.
__device__ __forceinline__ v8h cvt8(const float* __restrict__ p, int nt)
{
    const f32x4* q = (const f32x4*)p;
    f32x4 x, y;
    if (nt) { x = __builtin_nontemporal_load(q); y = __builtin_nontemporal_load(q + 1); }
    else    { x = q[0];                          y = q[1]; }
    v8h r;
    #pragma unroll
    for (int i = 0; i < 4; ++i) { r[i] = (_Float16)x[i]; r[i + 4] = (_Float16)y[i]; }
    return r;
}

__device__ __forceinline__ v8h ld8h(const _Float16* __restrict__ p)
{
    return *(const v8h*)p;   // 16B load
}

// ---------------------------------------------------------------------------
// WMMA GEMM v2:  C[M,N] = A @ Bt^T   (Bt is pre-packed f16 [N x K], rows = K)
//  - one wave computes a 16 x (16*NT) strip: A fragment reused NT times
//  - A: f32 row-major (optional row gather, optional non-temporal) or f16 (AF16)
//  - outputs: optional f32 C [M x N] (optional NT store) and/or
//             optional f16 transposed Ctr [N x M] (feeds next GEMM's B operand)
//  - gridDim.y batches independent problems (per-snapshot strides)
// Fragment layouts per CDNA5 ISA 7.12.2.
// ---------------------------------------------------------------------------
template <int NT, bool AF16>
__global__ __launch_bounds__(256)
void gemm2_kernel(const float* __restrict__ A, const _Float16* __restrict__ Ah,
                  const _Float16* __restrict__ Bt,
                  float* __restrict__ C, _Float16* __restrict__ Ctr,
                  const int* __restrict__ idxA, int mulA, int addA, long long strideIdx,
                  int M, int Nn, int Kd, int lda, int ldc, int ldctr,
                  long long strideA, long long strideB, long long strideC, long long strideCtr,
                  const float* __restrict__ bias, float scale, int relu, int ntA, int ntC)
{
    const long long bz = blockIdx.y;
    const float*    Ab  = A   ? A   + bz * strideA   : nullptr;
    const _Float16* Ahb = Ah  ? Ah  + bz * strideA   : nullptr;
    const _Float16* Btb = Bt  + bz * strideB;
    float*          Cb  = C   ? C   + bz * strideC   : nullptr;
    _Float16*       Ctb = Ctr ? Ctr + bz * strideCtr : nullptr;
    const int*      idx = idxA ? idxA + bz * strideIdx : nullptr;

    const int wave = threadIdx.x >> 5;
    const int lane = threadIdx.x & 31;
    const int tilesM  = M >> 4;
    const int nGroups = Nn / (16 * NT);
    const int strip   = blockIdx.x * 8 + wave;
    if (strip >= tilesM * nGroups) return;        // wave-uniform -> EXEC all-1s
    const int tm = strip / nGroups;
    const int tg = strip - tm * nGroups;

    const int mrow = lane & 15;
    const int hi   = lane >> 4;

    const float*    Arow  = nullptr;
    const _Float16* ArowH = nullptr;
    {
        int ra = tm * 16 + mrow;
        if (!AF16 && idx) ra = idx[ra] * mulA + addA;
        if (AF16) ArowH = Ahb + (size_t)ra * lda;
        else      Arow  = Ab  + (size_t)ra * lda;
    }

    const _Float16* Brow[NT];
    int ncol[NT];
    #pragma unroll
    for (int j = 0; j < NT; ++j) {
        ncol[j] = (tg * NT + j) * 16 + mrow;
        Brow[j] = Btb + (size_t)ncol[j] * Kd;
    }

    v8f zero = {};
    v8f acc[NT];
    #pragma unroll
    for (int j = 0; j < NT; ++j) acc[j] = zero;

    for (int k0 = 0; k0 < Kd; k0 += 32) {
        v16h a;
        if (AF16) {
            v8h lo = ld8h(ArowH + k0 + hi * 8);
            v8h h2 = ld8h(ArowH + k0 + 16 + hi * 8);
            a = __builtin_shufflevector(lo, h2, 0,1,2,3,4,5,6,7,8,9,10,11,12,13,14,15);
        } else {
            if (k0 + 128 < Kd) __builtin_prefetch(Arow + k0 + 128, 0, 1);
            v8h lo = cvt8(Arow + k0 + hi * 8, ntA);
            v8h h2 = cvt8(Arow + k0 + 16 + hi * 8, ntA);
            a = __builtin_shufflevector(lo, h2, 0,1,2,3,4,5,6,7,8,9,10,11,12,13,14,15);
        }
        #pragma unroll
        for (int j = 0; j < NT; ++j) {
            v8h lo = ld8h(Brow[j] + k0 + hi * 8);
            v8h h2 = ld8h(Brow[j] + k0 + 16 + hi * 8);
            v16h b = __builtin_shufflevector(lo, h2, 0,1,2,3,4,5,6,7,8,9,10,11,12,13,14,15);
            acc[j] = __builtin_amdgcn_wmma_f32_16x16x32_f16(
                         false, a, false, b, (short)0, acc[j], false, false);
        }
    }

    #pragma unroll
    for (int j = 0; j < NT; ++j) {
        const float bval = bias ? bias[ncol[j]] : 0.0f;
        if (Cb) {
            #pragma unroll
            for (int r = 0; r < 8; ++r) {         // C/D layout: VGPR r -> M = r + 8*hi
                const int mm = tm * 16 + r + hi * 8;
                float v = (acc[j][r] + bval) * scale;
                if (relu) v = fmaxf(v, 0.0f);
                const size_t off = (size_t)mm * ldc + ncol[j];
                if (ntC) __builtin_nontemporal_store(v, Cb + off);
                else     Cb[off] = v;
            }
        }
        if (Ctb) {
            v8h hv;
            #pragma unroll
            for (int r = 0; r < 8; ++r) {
                float v = (acc[j][r] + bval) * scale;
                if (relu) v = fmaxf(v, 0.0f);
                hv[r] = (_Float16)v;
            }
            *(v8h*)(Ctb + (size_t)ncol[j] * ldctr + tm * 16 + hi * 8) = hv;
        }
    }
}

// ---------------------------------------------------------------------------
// Pack f32 B[K,N] -> f16 Bt[N,K] (K-contiguous rows for WMMA B fragments).
__global__ void pack_bt_kernel(const float* __restrict__ B, _Float16* __restrict__ Bt,
                               int K, int N)
{
    int i = blockIdx.x * blockDim.x + threadIdx.x;
    if (i >= K * N) return;
    int n = i % N, k = i / N;
    Bt[(size_t)n * K + k] = (_Float16)B[i];
}

// Gather O rows for snapshot s and convert: Zh[s,i,k] = f16(O[(nodes[s,i]*S+s)*HID+k])
__global__ void pack_gather_kernel(const float* __restrict__ O,
                                   const int* __restrict__ nodes,
                                   _Float16* __restrict__ Zh)
{
    const int total = S_SNAP * NSUB * HID;
    int i = blockIdx.x * blockDim.x + threadIdx.x;
    if (i >= total) return;
    int k    = i & (HID - 1);
    int rest = i >> 6;
    int ii   = rest % NSUB;
    int s    = rest / NSUB;
    int n    = nodes[s * NSUB + ii];
    Zh[i] = (_Float16)O[((size_t)n * S_SNAP + s) * HID + k];
}

__global__ void zero_kernel(float* __restrict__ p, long long n)
{
    long long i = (long long)blockIdx.x * blockDim.x + threadIdx.x;
    if (i < n) p[i] = 0.0f;
}

// Scatter H2[s,i,:] -> X[nodes[s,i], s, :]
__global__ void scatter_kernel(const float* __restrict__ H2,
                               const int* __restrict__ nodes,
                               float* __restrict__ X)
{
    const int total = S_SNAP * NSUB * (HID / 4);
    int idx = blockIdx.x * blockDim.x + threadIdx.x;
    if (idx >= total) return;
    int c4   = idx & (HID / 4 - 1);
    int rest = idx >> 4;
    int i    = rest % NSUB;
    int s    = rest / NSUB;
    int n    = nodes[s * NSUB + i];
    const f32x4* src = (const f32x4*)(H2 + ((size_t)s * NSUB + i) * HID) + c4;
    f32x4*       dst = (f32x4*)(X + ((size_t)n * S_SNAP + s) * HID) + c4;
    *dst = *src;
}

// Per-node multi-head attention over S=8 snapshots; block = 64 threads = 1 node.
__global__ __launch_bounds__(64)
void attn_kernel(const float* __restrict__ Q, const float* __restrict__ K,
                 const float* __restrict__ V, const float* __restrict__ time_emb,
                 float* __restrict__ O)
{
    __shared__ float sq[S_SNAP][HID];
    __shared__ float sk[S_SNAP][HID];
    __shared__ float sv[S_SNAP][HID];

    const int n = blockIdx.x;
    const int t = threadIdx.x;
    const size_t base = (size_t)n * S_SNAP * HID;
    for (int i = t; i < S_SNAP * HID; i += 64) {
        sq[i / HID][i % HID] = Q[base + i];
        sk[i / HID][i % HID] = K[base + i];
        sv[i / HID][i % HID] = V[base + i];
    }
    __syncthreads();

    const int s = t >> 3;
    const int h = t & 7;

    float sc[S_SNAP];
    float mx = -3.4e38f;
    #pragma unroll
    for (int k = 0; k < S_SNAP; ++k) {
        float d = 0.0f;
        #pragma unroll
        for (int e = 0; e < DH; ++e)
            d += sq[s][h * DH + e] * sk[k][h * DH + e];
        d += time_emb[(k - s + S_SNAP - 1) * NH + h];
        sc[k] = d;
        mx = fmaxf(mx, d);
    }
    float sum = 0.0f;
    #pragma unroll
    for (int k = 0; k < S_SNAP; ++k) { sc[k] = __expf(sc[k] - mx); sum += sc[k]; }
    const float inv = 1.0f / sum;

    #pragma unroll
    for (int e = 0; e < DH; ++e) {
        float o = 0.0f;
        #pragma unroll
        for (int k = 0; k < S_SNAP; ++k) o += sc[k] * sv[k][h * DH + e];
        O[base + (size_t)s * HID + h * DH + e] = o * inv;
    }
}

// ---------------------------------------------------------------------------
extern "C" void kernel_launch(void* const* d_in, const int* in_sizes, int n_in,
                              void* d_out, int out_size, void* d_ws, size_t ws_size,
                              hipStream_t stream)
{
    const int*   nodes = (const int*)  d_in[0];
    const float* adj   = (const float*)d_in[1];
    const float* emb   = (const float*)d_in[2];
    const float* W1    = (const float*)d_in[3];
    const float* b1    = (const float*)d_in[4];
    const float* W2    = (const float*)d_in[5];
    const float* b2    = (const float*)d_in[6];
    const float* temb  = (const float*)d_in[7];
    const float* Wq    = (const float*)d_in[8];
    const float* bq    = (const float*)d_in[9];
    const float* Wk    = (const float*)d_in[10];
    const float* bk    = (const float*)d_in[11];
    const float* Wv    = (const float*)d_in[12];
    const float* bv    = (const float*)d_in[13];
    const float* Wo    = (const float*)d_in[14];
    const float* bo    = (const float*)d_in[15];
    float*       out   = (float*)d_out;

    const size_t nA = (size_t)S_SNAP * NSUB * HID;     // 2,048,000
    const size_t nX = (size_t)N_TOT * S_SNAP * HID;    // 10,240,000

    // f32 region
    float* H1b = (float*)d_ws;          // [S, NSUB, HID] (H1, then H2)
    float* X   = H1b + nA;              // [N, S, HID]; reused as attention output
    float* Qb  = X + nX;                // Q; reused for final projection O
    float* Kb  = Qb + nX;
    float* Vb  = Kb + nX;
    // f16 region
    _Float16* Tt  = (_Float16*)(Vb + nX);   // [S, HID, NSUB] (T1t, then T2t)
    _Float16* Zh  = Tt + nA;                // [S, NSUB, HID] gathered final rows
    _Float16* W1t = Zh + nA;                // [HID, F_IN]
    _Float16* W2t = W1t + (size_t)HID * F_IN;
    _Float16* Wqt = W2t + (size_t)HID * HID;
    _Float16* Wkt = Wqt + (size_t)HID * HID;
    _Float16* Wvt = Wkt + (size_t)HID * HID;
    _Float16* Wot = Wvt + (size_t)HID * HID;

    auto pack = [&](const float* B, _Float16* Bt, int K, int N) {
        int total = K * N;
        pack_bt_kernel<<<dim3((total + 255) / 256), dim3(256), 0, stream>>>(B, Bt, K, N);
    };
    // NT=4 GEMM (all 64-wide problems), batched over gridDim.y
    auto g4 = [&](const float* A, const int* idxA, long long strideIdx, int mulA, int addA,
                  long long strideA, int lda,
                  const _Float16* Bt, long long strideB,
                  float* C, long long strideC, int ldc,
                  _Float16* Ctr, long long strideCtr, int ldctr,
                  int M, int Nn, int Kd, const float* bias, float scale,
                  int relu, int batch, int ntA, int ntC) {
        int strips = (M / 16) * (Nn / 64);
        dim3 grid((strips + 7) / 8, batch);
        gemm2_kernel<4, false><<<grid, dim3(256), 0, stream>>>(
            A, nullptr, Bt, C, Ctr, idxA, mulA, addA, strideIdx,
            M, Nn, Kd, lda, ldc, ldctr, strideA, strideB, strideC, strideCtr,
            bias, scale, relu, ntA, ntC);
    };

    // ---- weight packs (tiny, once per call) -------------------------------
    pack(W1, W1t, F_IN, HID);
    pack(W2, W2t, HID, HID);
    pack(Wq, Wqt, HID, HID);
    pack(Wk, Wkt, HID, HID);
    pack(Wv, Wvt, HID, HID);
    pack(Wo, Wot, HID, HID);

    const long long sAdj = (long long)NSUB * NSUB;
    const long long sAH  = (long long)NSUB * HID;

    // ---- 2-layer GCN, batched over snapshots ------------------------------
    // T1t = (emb[nodes] @ W1)^T   (f16 packed, feeds next GEMM's B operand)
    g4(emb, nodes, NSUB, 1, 0, 0, F_IN, W1t, 0,
       nullptr, 0, 0, Tt, sAH, NSUB,
       NSUB, HID, F_IN, nullptr, 1.0f, 0, S_SNAP, 0, 0);
    // H1 = relu(adj @ T1 + b1)
    g4(adj, nullptr, 0, 0, 0, sAdj, NSUB, Tt, sAH,
       H1b, sAH, HID, nullptr, 0, 0,
       NSUB, HID, NSUB, b1, 1.0f, 1, S_SNAP, 1, 0);
    // T2t = (H1 @ W2)^T
    g4(H1b, nullptr, 0, 0, 0, sAH, HID, W2t, 0,
       nullptr, 0, 0, Tt, sAH, NSUB,
       NSUB, HID, HID, nullptr, 1.0f, 0, S_SNAP, 0, 0);
    // H2 = relu(adj @ T2 + b2)   (overwrites H1 buffer)
    g4(adj, nullptr, 0, 0, 0, sAdj, NSUB, Tt, sAH,
       H1b, sAH, HID, nullptr, 0, 0,
       NSUB, HID, NSUB, b2, 1.0f, 1, S_SNAP, 1, 0);

    // ---- scatter into dense [N, S, HID] -----------------------------------
    zero_kernel<<<dim3((unsigned)((nX + 255) / 256)), dim3(256), 0, stream>>>(
        X, (long long)nX);
    {
        int total = S_SNAP * NSUB * (HID / 4);
        scatter_kernel<<<dim3((total + 255) / 256), dim3(256), 0, stream>>>(
            H1b, nodes, X);
    }

    // ---- Q,K,V projections -------------------------------------------------
    const float qscale = 1.0f / sqrtf((float)DH);
    g4(X, nullptr, 0, 0, 0, 0, HID, Wqt, 0, Qb, 0, HID, nullptr, 0, 0,
       N_TOT * S_SNAP, HID, HID, bq, qscale, 0, 1, 0, 0);
    g4(X, nullptr, 0, 0, 0, 0, HID, Wkt, 0, Kb, 0, HID, nullptr, 0, 0,
       N_TOT * S_SNAP, HID, HID, bk, 1.0f, 0, 1, 0, 0);
    g4(X, nullptr, 0, 0, 0, 0, HID, Wvt, 0, Vb, 0, HID, nullptr, 0, 0,
       N_TOT * S_SNAP, HID, HID, bv, 1.0f, 0, 1, 0, 0);

    // ---- per-node attention (writes into X) -------------------------------
    attn_kernel<<<dim3(N_TOT), dim3(64), 0, stream>>>(Qb, Kb, Vb, temb, X);

    // ---- output projection O = attn @ Wo + bo (into Qb) -------------------
    g4(X, nullptr, 0, 0, 0, 0, HID, Wot, 0, Qb, 0, HID, nullptr, 0, 0,
       N_TOT * S_SNAP, HID, HID, bo, 1.0f, 0, 1, 0, 0);

    // ---- gather + f16-pack final rows -------------------------------------
    {
        int total = S_SNAP * NSUB * HID;
        pack_gather_kernel<<<dim3((total + 255) / 256), dim3(256), 0, stream>>>(
            Qb, nodes, Zh);
    }

    // ---- structure decoder: out[s] = Z Z^T (NT=2, f16 A, NT stores) -------
    {
        int strips = (NSUB / 16) * (NSUB / 32);
        dim3 grid((strips + 7) / 8, S_SNAP);
        gemm2_kernel<2, true><<<grid, dim3(256), 0, stream>>>(
            nullptr, Zh, Zh, out, nullptr, nullptr, 0, 0, 0,
            NSUB, NSUB, HID, HID, NSUB, 0,
            (long long)sAH, (long long)sAH, (long long)NSUB * NSUB, 0,
            nullptr, 1.0f, 0, 0, 1);
    }
}
